// Attention_58025008169314
// MI455X (gfx1250) — matrix operand order repaired and checked by reading, not compiled, not run on hardware
//
#include <hip/hip_runtime.h>
#include <math.h>

#define T_TOK   2048
#define NHEADS  32
#define HKV     8
#define DHEAD   128
#define NSAL    512
#define NSEQ    4
#define KVDIM   (HKV * DHEAD)     // 1024
#define QDIM    (NHEADS * DHEAD)  // 4096
#define SCALE_F 0.08838834764831845f

typedef __bf16 v16bf __attribute__((ext_vector_type(16)));
typedef float  v8f   __attribute__((ext_vector_type(8)));

union Frag {
  v16bf   bf;
  unsigned u[8];
  uint4    q4[2];
};

// float -> bf16 (round-to-nearest-even), manual so it always compiles
__device__ inline unsigned short f2bf(float f) {
  unsigned u = __float_as_uint(f);
  u += 0x7FFFu + ((u >> 16) & 1u);
  return (unsigned short)(u >> 16);
}
__device__ inline unsigned pk2(float lo, float hi) {
  return ((unsigned)f2bf(hi) << 16) | (unsigned)f2bf(lo);
}

// ---------------------------------------------------------------- prep kernels
__global__ __launch_bounds__(256) void init_salmap_k(int* salmap) {
  int t = blockIdx.x * 256 + threadIdx.x;
  if (t < T_TOK) salmap[t] = -1;
}
__global__ __launch_bounds__(256) void set_salmap_k(const long long* __restrict__ idx,
                                                    int* __restrict__ salmap) {
  int t = blockIdx.x * 256 + threadIdx.x;
  if (t < NSAL) salmap[(int)idx[t]] = t;
}
__global__ __launch_bounds__(256) void conv_k_k(const float* __restrict__ k,
                                                unsigned short* __restrict__ kbf) {
  int t = blockIdx.x * 256 + threadIdx.x;
  if (t < T_TOK * KVDIM) kbf[t] = f2bf(k[t]);
}
// Build transposed (d-major) bf16 value tensors:
//   vdT[(hkv*D+d)*T + key] = bf16(v_delta_sparse[key][hkv][d])
//   vnT[(hkv*D+d)*T + key] = bf16(v_cache_new  [key][hkv][d])
__global__ __launch_bounds__(256) void build_vT_k(const float* __restrict__ v,
                                                  const float* __restrict__ vcache,
                                                  const int* __restrict__ salmap,
                                                  unsigned short* __restrict__ vdT,
                                                  unsigned short* __restrict__ vnT) {
  int t = blockIdx.x * 256 + threadIdx.x;
  if (t >= KVDIM * T_TOK) return;
  int key = t & (T_TOK - 1);
  int hd  = t >> 11;                  // T_TOK == 2^11
  float c = vcache[(size_t)key * KVDIM + hd];
  int j = salmap[key];
  float vd = 0.0f, vn = c;
  if (j >= 0) {
    float vv = v[(size_t)j * KVDIM + hd];
    vd = vv - c;
    vn = vv;
  }
  vdT[t] = f2bf(vd);
  vnT[t] = f2bf(vn);
}

// --------------------------------------------------- softmax + P^T construction
// Per-lane state owns query column q = lane&15 of the S^T tiles.
__device__ inline void softmax_step(const v8f& Sa, const v8f& Sb, int hi,
                                    float& m, float& Z, v8f* acc, Frag& Bp) {
  float sa[8], sb[8];
  float mloc = -__builtin_inff();
#pragma unroll
  for (int r = 0; r < 8; ++r) {
    sa[r] = Sa[r] * SCALE_F;
    sb[r] = Sb[r] * SCALE_F;
    mloc = fmaxf(mloc, fmaxf(sa[r], sb[r]));
  }
  mloc = fmaxf(mloc, __shfl_xor(mloc, 16));
  float mnew = fmaxf(m, mloc);
  float corr = __expf(m - mnew);
  float ea[8], eb[8], sum = 0.f;
#pragma unroll
  for (int r = 0; r < 8; ++r) {
    ea[r] = __expf(sa[r] - mnew);
    eb[r] = __expf(sb[r] - mnew);
    sum += ea[r] + eb[r];
  }
  sum += __shfl_xor(sum, 16);
  Z = Z * corr + sum;
  m = mnew;
#pragma unroll
  for (int c = 0; c < 8; ++c) acc[c] *= corr;

  unsigned pa[4], pb[4];
#pragma unroll
  for (int v2 = 0; v2 < 4; ++v2) {
    pa[v2] = pk2(ea[2 * v2], ea[2 * v2 + 1]);
    pb[v2] = pk2(eb[2 * v2], eb[2 * v2 + 1]);
  }
#pragma unroll
  for (int v2 = 0; v2 < 4; ++v2) {
    unsigned qa = (unsigned)__shfl_xor((int)pa[v2], 16);
    unsigned qb = (unsigned)__shfl_xor((int)pb[v2], 16);
    Bp.u[v2]     = hi ? qb : pa[v2];       // keys 0..7   | 16..23
    Bp.u[4 + v2] = hi ? pb[v2] : qa;       // keys 8..15  | 24..31
  }
}

// ------------------------------------------------------- flash attention core
// One wave handles TWO 16-query tiles so every K/V A-fragment load feeds two
// WMMAs (32 WMMAs per 32 b128 loads per 32-key block).
// MODE 0: delta attention, out[row] = c_cache[row] + P(all same-seg keys) * vdT
// MODE 1: salient attention, rows gathered via idx, out[row] = P * vnT
template <int MODE>
__global__ __launch_bounds__(32)
void attn_kernel(const float* __restrict__ q,
                 const unsigned short* __restrict__ kbf,
                 const unsigned short* __restrict__ vT,
                 const float* __restrict__ c_cache,
                 const long long* __restrict__ idx,
                 const long long* __restrict__ cu,
                 float* __restrict__ out) {
  const int lane = threadIdx.x;
  const int lq   = lane & 15;   // query column this lane owns in S^T / acc
  const int hi   = lane >> 4;   // half-wave group
  const int tile = blockIdx.x;  // pair of 16-query tiles
  const int h    = blockIdx.y;
  const int hkv  = h >> 2;      // GQA: n_rep = 4
  const int aoff = hi ? 8 : 0;

  int row[2];
  if (MODE == 0) {
    row[0] = tile * 32 + lq;
    row[1] = row[0] + 16;
  } else {
    row[0] = (int)idx[tile * 32 + lq];
    row[1] = (int)idx[tile * 32 + 16 + lq];
  }

  // segment bounds (both tiles live in the same segment for this workload)
  int kbeg = 0, kend = T_TOK;
#pragma unroll
  for (int s = 0; s < NSEQ; ++s) {
    int a = (int)cu[s], b = (int)cu[s + 1];
    if (row[0] >= a && row[0] < b) { kbeg = a; kend = b; }
  }

  // B operands = Q^T : lane holds column q=lq, head-dim values 32*st + 8*hi + {0..7, 16..23}.
  Frag bq[2][4];
#pragma unroll
  for (int t2 = 0; t2 < 2; ++t2) {
    const float* qrow = q + (size_t)row[t2] * QDIM + (size_t)h * DHEAD;
#pragma unroll
    for (int st = 0; st < 4; ++st) {
      const float4* p = (const float4*)(qrow + st * 32 + (hi ? 8 : 0));
      float4 f0 = p[0], f1 = p[1], f2 = p[4], f3 = p[5];
      bq[t2][st].u[0] = pk2(f0.x, f0.y); bq[t2][st].u[1] = pk2(f0.z, f0.w);
      bq[t2][st].u[2] = pk2(f1.x, f1.y); bq[t2][st].u[3] = pk2(f1.z, f1.w);
      bq[t2][st].u[4] = pk2(f2.x, f2.y); bq[t2][st].u[5] = pk2(f2.z, f2.w);
      bq[t2][st].u[6] = pk2(f3.x, f3.y); bq[t2][st].u[7] = pk2(f3.z, f3.w);
    }
  }

  const v8f vzero = {0.f, 0.f, 0.f, 0.f, 0.f, 0.f, 0.f, 0.f};
  v8f acc0[8], acc1[8];             // out^T per tile, 8 d-chunks of 16
#pragma unroll
  for (int c = 0; c < 8; ++c) { acc0[c] = vzero; acc1[c] = vzero; }
  float m0 = -__builtin_inff(), Z0 = 0.0f;
  float m1 = -__builtin_inff(), Z1 = 0.0f;

  const unsigned short* vrow0 = vT + ((size_t)(hkv * DHEAD) + lq) * T_TOK;

  for (int kb = kbeg; kb < kend; kb += 32) {
    // ---- S^T = K * Q^T  (two 16x16 key tiles x two query tiles)
    const unsigned short* ka  = kbf + (size_t)(kb + lq) * KVDIM + (size_t)hkv * DHEAD;
    const unsigned short* kbp = ka + (size_t)16 * KVDIM;
    v8f Sa0 = vzero, Sb0 = vzero, Sa1 = vzero, Sb1 = vzero;
#pragma unroll
    for (int st = 0; st < 4; ++st) {
      Frag Aa, Ab;
      int off = st * 32 + aoff;
      Aa.q4[0] = *(const uint4*)(ka + off);
      Aa.q4[1] = *(const uint4*)(ka + off + 16);
      Ab.q4[0] = *(const uint4*)(kbp + off);
      Ab.q4[1] = *(const uint4*)(kbp + off + 16);
      Sa0 = __builtin_amdgcn_wmma_f32_16x16x32_bf16(false, Aa.bf, false, bq[0][st].bf,
                                                    (short)0, Sa0, false, false);
      Sb0 = __builtin_amdgcn_wmma_f32_16x16x32_bf16(false, Ab.bf, false, bq[0][st].bf,
                                                    (short)0, Sb0, false, false);
      Sa1 = __builtin_amdgcn_wmma_f32_16x16x32_bf16(false, Aa.bf, false, bq[1][st].bf,
                                                    (short)0, Sa1, false, false);
      Sb1 = __builtin_amdgcn_wmma_f32_16x16x32_bf16(false, Ab.bf, false, bq[1][st].bf,
                                                    (short)0, Sb1, false, false);
    }

    // ---- online softmax + P^T for both query tiles
    Frag Bp0, Bp1;
    softmax_step(Sa0, Sb0, hi, m0, Z0, acc0, Bp0);
    softmax_step(Sa1, Sb1, hi, m1, Z1, acc1, Bp1);

    // ---- out^T += V^T * P^T   (A = V^T chunk, shared across both tiles)
#pragma unroll
    for (int c = 0; c < 8; ++c) {
      Frag Av;
      const unsigned short* vr = vrow0 + (size_t)(16 * c) * T_TOK + kb;
      Av.q4[0] = *(const uint4*)(vr + 2 * aoff);
      Av.q4[1] = *(const uint4*)(vr + 2 * aoff + 8);
      acc0[c] = __builtin_amdgcn_wmma_f32_16x16x32_bf16(false, Av.bf, false, Bp0.bf,
                                                        (short)0, acc0[c], false, false);
      acc1[c] = __builtin_amdgcn_wmma_f32_16x16x32_bf16(false, Av.bf, false, Bp1.bf,
                                                        (short)0, acc1[c], false, false);
    }
  }

  // ---- epilogue: lane owns query rows row[0]/row[1], d = 16c + 8*hi + r
#pragma unroll
  for (int t2 = 0; t2 < 2; ++t2) {
    const v8f* acc = t2 ? acc1 : acc0;
    float invZ = 1.0f / (t2 ? Z1 : Z0);
    size_t base = (size_t)row[t2] * QDIM + (size_t)h * DHEAD;
#pragma unroll
    for (int c = 0; c < 8; ++c) {
      int d0 = 16 * c + hi * 8;
      float4 o0, o1;
      o0.x = acc[c][0] * invZ; o0.y = acc[c][1] * invZ;
      o0.z = acc[c][2] * invZ; o0.w = acc[c][3] * invZ;
      o1.x = acc[c][4] * invZ; o1.y = acc[c][5] * invZ;
      o1.z = acc[c][6] * invZ; o1.w = acc[c][7] * invZ;
      if (MODE == 0) {
        float4 c0 = *(const float4*)(c_cache + base + d0);
        float4 c1 = *(const float4*)(c_cache + base + d0 + 4);
        o0.x += c0.x; o0.y += c0.y; o0.z += c0.z; o0.w += c0.w;
        o1.x += c1.x; o1.y += c1.y; o1.z += c1.z; o1.w += c1.w;
      }
      *(float4*)(out + base + d0)     = o0;
      *(float4*)(out + base + d0 + 4) = o1;
    }
  }
}

// ------------------------------------------------------------- cosine similarity
__global__ __launch_bounds__(256)
void cos_kernel(const float* __restrict__ cc, const float* __restrict__ nc,
                float* __restrict__ outcos) {
  int row = blockIdx.x;
  int t = threadIdx.x;
  const float* a = cc + (size_t)row * QDIM;
  const float* b = nc + (size_t)row * QDIM;
  float sn = 0.f, s0 = 0.f, s1 = 0.f;
  for (int e = t; e < QDIM; e += 256) {
    float x = a[e], y = b[e];
    sn += x * y; s0 += x * x; s1 += y * y;
  }
#pragma unroll
  for (int o = 16; o >= 1; o >>= 1) {
    sn += __shfl_xor(sn, o);
    s0 += __shfl_xor(s0, o);
    s1 += __shfl_xor(s1, o);
  }
  __shared__ float red[3][8];
  int w = t >> 5;
  if ((t & 31) == 0) { red[0][w] = sn; red[1][w] = s0; red[2][w] = s1; }
  __syncthreads();
  if (t == 0) {
    float N = 0.f, C = 0.f, D = 0.f;
    for (int i = 0; i < 8; ++i) { N += red[0][i]; C += red[1][i]; D += red[2][i]; }
    outcos[row] = N / (sqrtf(C) * sqrtf(D) + 1e-8f);
  }
}

// --------------------------------------------------------------------- launcher
extern "C" void kernel_launch(void* const* d_in, const int* in_sizes, int n_in,
                              void* d_out, int out_size, void* d_ws, size_t ws_size,
                              hipStream_t stream) {
  (void)in_sizes; (void)n_in; (void)out_size; (void)ws_size;
  const float*     q      = (const float*)d_in[0];
  const float*     k      = (const float*)d_in[1];
  const float*     v      = (const float*)d_in[2];
  const float*     vcache = (const float*)d_in[3];
  const float*     ccache = (const float*)d_in[4];
  const long long* idx    = (const long long*)d_in[5];
  const long long* cu     = (const long long*)d_in[6];
  float* out = (float*)d_out;

  char* ws = (char*)d_ws;
  const size_t SZ_KV = (size_t)T_TOK * KVDIM * sizeof(unsigned short);  // 4 MB
  unsigned short* kbf = (unsigned short*)(ws);
  unsigned short* vdT = (unsigned short*)(ws + SZ_KV);
  unsigned short* vnT = (unsigned short*)(ws + 2 * SZ_KV);
  int*            sal = (int*)(ws + 3 * SZ_KV);

  init_salmap_k<<<T_TOK / 256, 256, 0, stream>>>(sal);
  set_salmap_k<<<(NSAL + 255) / 256, 256, 0, stream>>>(idx, sal);
  conv_k_k<<<(T_TOK * KVDIM) / 256, 256, 0, stream>>>(k, kbf);
  build_vT_k<<<(KVDIM * T_TOK) / 256, 256, 0, stream>>>(v, vcache, sal, vdT, vnT);

  // delta attention: writes c_cache + delta_c for every row/head (2 tiles/wave)
  attn_kernel<0><<<dim3(T_TOK / 32, NHEADS), 32, 0, stream>>>(q, kbf, vdT, ccache,
                                                              idx, cu, out);
  // salient attention: overwrites salient rows with o_salient
  attn_kernel<1><<<dim3(NSAL / 32, NHEADS), 32, 0, stream>>>(q, kbf, vnT, ccache,
                                                             idx, cu, out);
  // cosine similarity vs old context
  cos_kernel<<<T_TOK, 256, 0, stream>>>(ccache, out, out + (size_t)T_TOK * QDIM);
}
